// ieHGCNConv_16475494547763
// MI455X (gfx1250) — matrix operand order repaired and checked by reading, not compiled, not run on hardware
//
#include <hip/hip_runtime.h>

#define N_NODES 50000
#define N_EDGES 262144

typedef __attribute__((ext_vector_type(16))) __bf16 v16bf;
typedef __attribute__((ext_vector_type(8)))  __bf16 v8bf;
typedef __attribute__((ext_vector_type(8)))  float  v8f;
typedef __attribute__((ext_vector_type(4)))  unsigned u32x4;
typedef __attribute__((ext_vector_type(8)))  int      i32x8;
typedef __attribute__((ext_vector_type(4)))  int      i32x4;

#if __has_builtin(__builtin_amdgcn_tensor_load_to_lds)
#define USE_TDM 1
#else
#define USE_TDM 0
#endif

// ---------- helpers ----------
__device__ __forceinline__ unsigned f2ord(float f) {
  unsigned u = __float_as_uint(f);
  return (u & 0x80000000u) ? ~u : (u | 0x80000000u);
}
__device__ __forceinline__ float ord2f(unsigned u) {
  unsigned v = (u & 0x80000000u) ? (u & 0x7fffffffu) : ~u;
  return __uint_as_float(v);
}
__device__ __forceinline__ float wave_sum(float v) {
#pragma unroll
  for (int o = 16; o > 0; o >>= 1) v += __shfl_xor(v, o, 32);
  return v;
}
__device__ __forceinline__ float elu1(float x)  { return x > 0.f ? x : (__expf(x) - 1.f); }
__device__ __forceinline__ float lrelu(float x) { return x > 0.f ? x : 0.2f * x; }

// ---------- GEMM: C[M x 256] = A[M x 256] @ W[256 x 256] + bias, bf16 WMMA ----------
// block = 256 threads (8 waves); each block computes 64 rows x 256 cols.
// Wave w owns N-tiles {2w, 2w+1} and M-tiles 0..3  ->  8 WMMAs / K-step / wave.
// A tile staged via Tensor Data Mover (TDM) when available; W panel staged into a
// fragment-native LDS swizzle so every B fragment is two ds_load_b128.
__global__ __launch_bounds__(256)
void gemm256_wmma_bf16(const float* __restrict__ A, const float* __restrict__ W,
                       const float* __restrict__ bias, float* __restrict__ C) {
  __shared__ __align__(16) __bf16 As[64 * 32];          // [m][k]   4 KB
  __shared__ __align__(16) __bf16 Bs[16 * 32 * 16];     // [nt][lane][i] 16 KB
#if USE_TDM
  __shared__ __align__(16) float Af32[64 * 32];         // TDM landing pad, 8 KB
#endif
  const int tid  = threadIdx.x;
  const int lane = tid & 31;
  const int wave = tid >> 5;
  const int half = lane >> 4;
  const int ml   = lane & 15;
  const long mbase = (long)blockIdx.x * 64;

  // staging identity: this thread owns column n = tid of the W panel
  const int nt_s = tid >> 4, ml_s = tid & 15;
  // staging identity for A conversion: 8 consecutive k at (mA, kA)
  const int mA = tid >> 2, kA = (tid & 3) * 8;

  v8f acc[4][2];
#pragma unroll
  for (int mt = 0; mt < 4; ++mt)
#pragma unroll
    for (int j = 0; j < 2; ++j)
      acc[mt][j] = (v8f){0.f,0.f,0.f,0.f,0.f,0.f,0.f,0.f};

  for (int k0 = 0; k0 < 256; k0 += 32) {
#if USE_TDM
    if (wave == 0) {
      // 2D D#: 32 (k) x 64 (rows) tile of f32, row stride 256, OOB rows read 0
      unsigned lds_off = (unsigned)(unsigned long long)(size_t)(void*)&Af32[0];
      unsigned long long ga = (unsigned long long)(size_t)(const void*)(A + mbase * 256 + k0);
      unsigned rows = (mbase + 64 <= N_NODES) ? 64u : (unsigned)(N_NODES - mbase);
      u32x4 g0;
      g0[0] = 1u;                                           // count=1, user mode
      g0[1] = lds_off;                                      // lds_addr
      g0[2] = (unsigned)ga;                                 // global_addr[31:0]
      g0[3] = (unsigned)((ga >> 32) & 0x1FFFFFFull) | (2u << 30); // addr[56:32] | type=2
      i32x8 g1;
      g1[0] = (int)(2u << 16);                              // data_size = 4B
      g1[1] = (int)(256u << 16);                            // tensor_dim0 = 256
      g1[2] = (int)((rows & 0xFFFFu) << 16);                // tensor_dim1 lo16
      g1[3] = (int)(((rows >> 16) & 0xFFFFu) | (32u << 16)); // dim1 hi16 | tile_dim0=32
      g1[4] = 64;                                           // tile_dim1 = 64
      g1[5] = 256;                                          // tensor_dim0_stride = 256
      g1[6] = 0; g1[7] = 0;
      i32x4 gz = {0, 0, 0, 0};
#if __clang_major__ >= 23
      i32x8 gz8 = {0, 0, 0, 0, 0, 0, 0, 0};
      __builtin_amdgcn_tensor_load_to_lds(g0, g1, gz, gz, gz8, 0);
#else
      __builtin_amdgcn_tensor_load_to_lds(g0, g1, gz, gz, 0);
#endif
    }
#endif
    // W panel loads: column n=tid, 32 k values; coalesced b32 across each wave
    float w[32];
#pragma unroll
    for (int kk = 0; kk < 32; ++kk)
      w[kk] = W[(long)(k0 + kk) * 256 + tid];

#if !USE_TDM
    long rowA = mbase + mA; if (rowA >= N_NODES) rowA = N_NODES - 1;
    const float4 a0g = *(const float4*)(A + rowA * 256 + k0 + kA);
    const float4 a1g = *(const float4*)(A + rowA * 256 + k0 + kA + 4);
#endif

#if USE_TDM
    if (wave == 0) __builtin_amdgcn_s_wait_tensorcnt(0);
#endif
    __syncthreads();   // prev-iter fragment reads done; TDM tile visible to all waves

    { // W panel -> Bs in fragment-native layout: Bs[(nt*32 + half*16 + ml)*16 + i]
      v8bf q0, q1, q2, q3;
#pragma unroll
      for (int i = 0; i < 8; ++i) {
        q0[i] = (__bf16)w[i];      q1[i] = (__bf16)w[8 + i];
        q2[i] = (__bf16)w[16 + i]; q3[i] = (__bf16)w[24 + i];
      }
      const int b0 = (nt_s * 32 + ml_s) * 16;
      const int b1 = (nt_s * 32 + 16 + ml_s) * 16;
      *(v8bf*)(Bs + b0)     = q0;  *(v8bf*)(Bs + b0 + 8) = q1;
      *(v8bf*)(Bs + b1)     = q2;  *(v8bf*)(Bs + b1 + 8) = q3;
    }
    { // A tile -> As (bf16), 8 elems per thread
#if USE_TDM
      const float4 a0g = *(const float4*)(Af32 + mA * 32 + kA);
      const float4 a1g = *(const float4*)(Af32 + mA * 32 + kA + 4);
#endif
      v8bf pa;
      pa[0] = (__bf16)a0g.x; pa[1] = (__bf16)a0g.y; pa[2] = (__bf16)a0g.z; pa[3] = (__bf16)a0g.w;
      pa[4] = (__bf16)a1g.x; pa[5] = (__bf16)a1g.y; pa[6] = (__bf16)a1g.z; pa[7] = (__bf16)a1g.w;
      *(v8bf*)(As + mA * 32 + kA) = pa;
    }
    __syncthreads();   // As / Bs ready

    // B fragments for this wave's two N-tiles: two ds_load_b128 each
    const int nt0 = wave * 2, nt1 = wave * 2 + 1;
    const v8bf bl0 = *(const v8bf*)(Bs + (nt0 * 32 + lane) * 16);
    const v8bf bh0 = *(const v8bf*)(Bs + (nt0 * 32 + lane) * 16 + 8);
    const v8bf bl1 = *(const v8bf*)(Bs + (nt1 * 32 + lane) * 16);
    const v8bf bh1 = *(const v8bf*)(Bs + (nt1 * 32 + lane) * 16 + 8);
    const v16bf bf0 = __builtin_shufflevector(bl0, bh0, 0,1,2,3,4,5,6,7,8,9,10,11,12,13,14,15);
    const v16bf bf1 = __builtin_shufflevector(bl1, bh1, 0,1,2,3,4,5,6,7,8,9,10,11,12,13,14,15);

#pragma unroll
    for (int mt = 0; mt < 4; ++mt) {
      const v8bf al = *(const v8bf*)(As + (mt * 16 + ml) * 32 + half * 16);
      const v8bf ah = *(const v8bf*)(As + (mt * 16 + ml) * 32 + half * 16 + 8);
      const v16bf af = __builtin_shufflevector(al, ah, 0,1,2,3,4,5,6,7,8,9,10,11,12,13,14,15);
      acc[mt][0] = __builtin_amdgcn_wmma_f32_16x16x32_bf16(false, af, false, bf0,
                                                           (short)0, acc[mt][0], false, false);
      acc[mt][1] = __builtin_amdgcn_wmma_f32_16x16x32_bf16(false, af, false, bf1,
                                                           (short)0, acc[mt][1], false, false);
    }
    __syncthreads();   // fragments consumed before next stage overwrites LDS
  }

  const int col = wave * 32 + ml;
  const float b0 = bias[col], b1 = bias[col + 16];
#pragma unroll
  for (int mt = 0; mt < 4; ++mt) {
#pragma unroll
    for (int r = 0; r < 8; ++r) {
      long row = mbase + mt * 16 + r + 8 * half;
      if (row < N_NODES) {
        C[row * 256 + col]      = acc[mt][0][r] + b0;
        C[row * 256 + col + 16] = acc[mt][1][r] + b1;
      }
    }
  }
}

// ---------- collapse (Wk@Wal, Wq@War) into 256-vectors + scalars ----------
__global__ __launch_bounds__(256)
void prep_attvec(const float* __restrict__ Wq, const float* __restrict__ bq,
                 const float* __restrict__ Wk, const float* __restrict__ bk,
                 const float* __restrict__ Wal, const float* __restrict__ bal,
                 const float* __restrict__ War, const float* __restrict__ bar,
                 float* __restrict__ wl, float* __restrict__ wr,
                 float* __restrict__ cl, float* __restrict__ cr) {
  int t = blockIdx.x;      // node type 0/1
  int i = threadIdx.x;     // output channel 0..255
  float sl = 0.f, sr = 0.f;
  for (int j = 0; j < 64; ++j) {
    sl += Wk[(t * 256 + i) * 64 + j] * Wal[t * 64 + j];
    sr += Wq[(t * 256 + i) * 64 + j] * War[t * 64 + j];
  }
  wl[t * 256 + i] = sl;
  wr[t * 256 + i] = sr;
  if (i == 0) {
    float c1 = bal[t], c2 = bar[t];
    for (int j = 0; j < 64; ++j) {
      c1 += bk[t * 64 + j] * Wal[t * 64 + j];
      c2 += bq[t * 64 + j] * War[t * 64 + j];
    }
    cl[t] = c1; cr[t] = c2;
  }
}

// ---------- per-relation init: rst = gbias (broadcast), m = enc(-inf), den = 0 ----
__global__ __launch_bounds__(256)
void init_rel(float* __restrict__ rst, const float* __restrict__ gb,
              unsigned* __restrict__ m, float* __restrict__ den) {
  long i = (long)blockIdx.x * 256 + threadIdx.x;   // grid = N_NODES blocks
  rst[i] = gb[threadIdx.x];
  if (i < N_NODES) { m[i] = 0u; den[i] = 0.f; }
}

// ---------- pass 1: per-edge logit + segment max (ordered-uint atomicMax) -------
__global__ __launch_bounds__(256)
void edge_logit(const int* __restrict__ src, const int* __restrict__ dst,
                const float* __restrict__ fts, const float* __restrict__ ftd,
                const float* __restrict__ av, float* __restrict__ logit,
                unsigned* __restrict__ m) {
  int e = blockIdx.x * 8 + (threadIdx.x >> 5);
  int lane = threadIdx.x & 31;
  long so = (long)src[e] * 256 + lane * 8;
  long dO = (long)dst[e] * 256 + lane * 8;
  float4 s0 = *(const float4*)(fts + so);
  float4 s1 = *(const float4*)(fts + so + 4);
  float4 d0 = *(const float4*)(ftd + dO);
  float4 d1 = *(const float4*)(ftd + dO + 4);
  float4 a0 = *(const float4*)(av + lane * 8);
  float4 a1 = *(const float4*)(av + lane * 8 + 4);
  float s = a0.x * lrelu(s0.x + d0.x) + a0.y * lrelu(s0.y + d0.y)
          + a0.z * lrelu(s0.z + d0.z) + a0.w * lrelu(s0.w + d0.w)
          + a1.x * lrelu(s1.x + d1.x) + a1.y * lrelu(s1.y + d1.y)
          + a1.z * lrelu(s1.z + d1.z) + a1.w * lrelu(s1.w + d1.w);
  s = wave_sum(s);
  if (lane == 0) {
    logit[e] = s;
    atomicMax(&m[dst[e]], f2ord(s));
  }
}

// ---------- pass 2: exp + denominator ----------
__global__ __launch_bounds__(256)
void edge_exp(const int* __restrict__ dst, const float* __restrict__ logit,
              const unsigned* __restrict__ m, float* __restrict__ ex,
              float* __restrict__ den) {
  int e = blockIdx.x * 256 + threadIdx.x;
  if (e >= N_EDGES) return;
  int d = dst[e];
  float mm = ord2f(m[d]);
  if (!(mm > -3.0e38f && mm < 3.0e38f)) mm = 0.f;   // zero-in-degree safety
  float v = __expf(logit[e] - mm);
  ex[e] = v;
  atomicAdd(&den[d], v);
}

// ---------- pass 3: alpha * ft_src scatter-add into rst ----------
__global__ __launch_bounds__(256)
void edge_scatter(const int* __restrict__ src, const int* __restrict__ dst,
                  const float* __restrict__ fts, const float* __restrict__ ex,
                  const float* __restrict__ den, float* __restrict__ rst) {
  int e = blockIdx.x * 8 + (threadIdx.x >> 5);
  int lane = threadIdx.x & 31;
  int d = dst[e];
  float dn = den[d];
  float alpha = ex[e] / (dn > 0.f ? dn : 1.f);
  long so = (long)src[e] * 256 + lane * 8;
  long dO = (long)d * 256 + lane * 8;
  float4 s0 = *(const float4*)(fts + so);
  float4 s1 = *(const float4*)(fts + so + 4);
  atomicAdd(&rst[dO + 0], alpha * s0.x);
  atomicAdd(&rst[dO + 1], alpha * s0.y);
  atomicAdd(&rst[dO + 2], alpha * s0.z);
  atomicAdd(&rst[dO + 3], alpha * s0.w);
  atomicAdd(&rst[dO + 4], alpha * s1.x);
  atomicAdd(&rst[dO + 5], alpha * s1.y);
  atomicAdd(&rst[dO + 6], alpha * s1.z);
  atomicAdd(&rst[dO + 7], alpha * s1.w);
}

// ---------- final: h_l/h_r, elu gate, softmax-of-2, output elu ----------
__global__ __launch_bounds__(256)
void final_mix(const float* __restrict__ z, const float* __restrict__ rst,
               const float* __restrict__ wl, const float* __restrict__ wr,
               const float* __restrict__ cl, const float* __restrict__ cr,
               int t, float* __restrict__ out) {
  int n = blockIdx.x * 8 + (threadIdx.x >> 5);
  int lane = threadIdx.x & 31;
  long zo = (long)n * 256 + lane * 8;
  float s_zl = 0.f, s_zr = 0.f, s_rl = 0.f;
  float zz[8], rr[8];
#pragma unroll
  for (int j = 0; j < 8; ++j) {
    float w_l = wl[t * 256 + lane * 8 + j];
    float w_r = wr[t * 256 + lane * 8 + j];
    zz[j] = z[zo + j]; rr[j] = rst[zo + j];
    s_zl += zz[j] * w_l; s_zr += zz[j] * w_r; s_rl += rr[j] * w_l;
  }
  s_zl = wave_sum(s_zl); s_zr = wave_sum(s_zr); s_rl = wave_sum(s_rl);
  float hl = s_zl + cl[t], hr = s_zr + cr[t];
  float a_self = elu1(hl + hr);
  float e_att  = elu1(s_rl + cl[t] + hr);
  float mx = fmaxf(a_self, e_att);
  float ea = __expf(a_self - mx), eb = __expf(e_att - mx);
  float inv = 1.f / (ea + eb);
  float a0 = ea * inv, a1 = eb * inv;
#pragma unroll
  for (int j = 0; j < 8; ++j)
    out[zo + j] = elu1(a0 * zz[j] + a1 * rr[j]);
}

// ---------- host launch ----------
extern "C" void kernel_launch(void* const* d_in, const int* in_sizes, int n_in,
                              void* d_out, int out_size, void* d_ws, size_t ws_size,
                              hipStream_t stream) {
  (void)in_sizes; (void)n_in; (void)out_size; (void)ws_size;
  const float* hA     = (const float*)d_in[0];
  const float* hB     = (const float*)d_in[1];
  const float* Wself  = (const float*)d_in[2];
  const float* bself  = (const float*)d_in[3];
  const float* Wq     = (const float*)d_in[4];
  const float* bq     = (const float*)d_in[5];
  const float* Wk     = (const float*)d_in[6];
  const float* bk     = (const float*)d_in[7];
  const float* Wal    = (const float*)d_in[8];
  const float* bal    = (const float*)d_in[9];
  const float* War    = (const float*)d_in[10];
  const float* bar    = (const float*)d_in[11];
  const float* Wsrc   = (const float*)d_in[12];
  const float* bsrc   = (const float*)d_in[13];
  const float* Wdst   = (const float*)d_in[14];
  const float* bdst   = (const float*)d_in[15];
  const float* attn_v = (const float*)d_in[16];
  const float* gbias  = (const float*)d_in[17];
  const int* eab_src  = (const int*)d_in[18];
  const int* eab_dst  = (const int*)d_in[19];
  const int* eba_src  = (const int*)d_in[20];
  const int* eba_dst  = (const int*)d_in[21];

  const long NB = (long)N_NODES * 256;
  float* p = (float*)d_ws;
  float* z0       = p; p += NB;
  float* z1       = p; p += NB;
  float* fts_ab   = p; p += NB;
  float* ftd_ab   = p; p += NB;
  float* fts_ba   = p; p += NB;
  float* ftd_ba   = p; p += NB;
  float* rst_ab   = p; p += NB;
  float* rst_ba   = p; p += NB;
  float* logit_ab = p; p += N_EDGES;
  float* ex_ab    = p; p += N_EDGES;
  float* logit_ba = p; p += N_EDGES;
  float* ex_ba    = p; p += N_EDGES;
  unsigned* m_ab  = (unsigned*)p; p += N_NODES;
  float* den_ab   = p; p += N_NODES;
  unsigned* m_ba  = (unsigned*)p; p += N_NODES;
  float* den_ba   = p; p += N_NODES;
  float* wl       = p; p += 512;
  float* wr       = p; p += 512;
  float* cl       = p; p += 2;
  float* cr       = p; p += 2;

  float* outA = (float*)d_out;
  float* outB = outA + NB;

  dim3 blk(256);
  prep_attvec<<<2, blk, 0, stream>>>(Wq, bq, Wk, bk, Wal, bal, War, bar, wl, wr, cl, cr);

  const int GBLK = (N_NODES + 63) / 64;  // 782
  gemm256_wmma_bf16<<<GBLK, blk, 0, stream>>>(hA, Wself,         bself,       z0);
  gemm256_wmma_bf16<<<GBLK, blk, 0, stream>>>(hB, Wself + 65536, bself + 256, z1);
  gemm256_wmma_bf16<<<GBLK, blk, 0, stream>>>(hA, Wsrc,          bsrc,        fts_ab);
  gemm256_wmma_bf16<<<GBLK, blk, 0, stream>>>(hB, Wdst,          bdst,        ftd_ab);
  gemm256_wmma_bf16<<<GBLK, blk, 0, stream>>>(hB, Wsrc + 65536,  bsrc + 256,  fts_ba);
  gemm256_wmma_bf16<<<GBLK, blk, 0, stream>>>(hA, Wdst + 65536,  bdst + 256,  ftd_ba);

  init_rel<<<N_NODES, blk, 0, stream>>>(rst_ab, gbias,       m_ab, den_ab);  // etype 0
  init_rel<<<N_NODES, blk, 0, stream>>>(rst_ba, gbias + 256, m_ba, den_ba);  // etype 1

  edge_logit  <<<N_EDGES / 8,         blk, 0, stream>>>(eab_src, eab_dst, fts_ab, ftd_ab, attn_v,       logit_ab, m_ab);
  edge_exp    <<<(N_EDGES + 255)/256, blk, 0, stream>>>(eab_dst, logit_ab, m_ab, ex_ab, den_ab);
  edge_scatter<<<N_EDGES / 8,         blk, 0, stream>>>(eab_src, eab_dst, fts_ab, ex_ab, den_ab, rst_ab);

  edge_logit  <<<N_EDGES / 8,         blk, 0, stream>>>(eba_src, eba_dst, fts_ba, ftd_ba, attn_v + 256, logit_ba, m_ba);
  edge_exp    <<<(N_EDGES + 255)/256, blk, 0, stream>>>(eba_dst, logit_ba, m_ba, ex_ba, den_ba);
  edge_scatter<<<N_EDGES / 8,         blk, 0, stream>>>(eba_src, eba_dst, fts_ba, ex_ba, den_ba, rst_ba);

  final_mix<<<N_NODES / 8, blk, 0, stream>>>(z0, rst_ba, wl, wr, cl, cr, 0, outA);
  final_mix<<<N_NODES / 8, blk, 0, stream>>>(z1, rst_ab, wl, wr, cl, cr, 1, outB);
}
